// GAT_GCN_63685775065447
// MI455X (gfx1250) — matrix operand order repaired and checked by reading, not compiled, not run on hardware
//
#include <hip/hip_runtime.h>
#include <hip/hip_bf16.h>

// ---------------- constants from the reference ----------------
#define DDIM 64
#define HHEADS 10
#define HD 640           // HHEADS * DDIM
#define NEG_SLOPE 0.2f

typedef __attribute__((ext_vector_type(8)))  float  v8f;
typedef __attribute__((ext_vector_type(16))) __bf16 v16bf;

union Frag16 { unsigned int u[8]; v16bf v; };

// ---------------- helpers ----------------
__device__ __forceinline__ unsigned short f2bf(float f) {
    unsigned int u = __float_as_uint(f);
    u += 0x7FFFu + ((u >> 16) & 1u);     // round-to-nearest-even
    return (unsigned short)(u >> 16);
}
// monotone float<->uint map so atomicMax(uint) == float max (handles negatives)
__device__ __forceinline__ unsigned int fenc(float f) {
    unsigned int u = __float_as_uint(f);
    return (u & 0x80000000u) ? ~u : (u | 0x80000000u);
}
__device__ __forceinline__ float fdec(unsigned int k) {
    unsigned int u = (k & 0x80000000u) ? (k & 0x7FFFFFFFu) : ~k;
    return __uint_as_float(u);
}
#define ENC_NEG_INF 0x007FFFFFu   // fenc(-inf)
#define ENC_ZERO    0x80000000u   // fenc(0.0f)

// ---------------- small utility kernels ----------------
__global__ void fill_u32_kernel(unsigned int* __restrict__ p, unsigned int v, int n) {
    int i = blockIdx.x * blockDim.x + threadIdx.x;
    if (i < n) p[i] = v;
}
__global__ void cvt_bf16_rowmajor(const float* __restrict__ src,
                                  unsigned short* __restrict__ dst, int n) {
    int i = blockIdx.x * blockDim.x + threadIdx.x;
    if (i < n) dst[i] = f2bf(src[i]);
}
// src is KxNcols row-major; dst is Ncols x K (B transposed, K contiguous)
__global__ void cvt_bf16_transpose(const float* __restrict__ src,
                                   unsigned short* __restrict__ dst, int K, int Ncols) {
    int i = blockIdx.x * blockDim.x + threadIdx.x;
    if (i >= K * Ncols) return;
    int k = i / Ncols, n = i % Ncols;
    dst[(size_t)n * K + k] = f2bf(src[i]);
}

// ---------------- bf16 WMMA GEMM: C[MxNcols] = A[MxK] * B[KxNcols] ----------------
// A: row-major bf16. Bt: B transposed (Ncols x K) bf16. One wave -> 16x64 strip.
__global__ void gemm_bf16_wmma(const unsigned short* __restrict__ A,
                               const unsigned short* __restrict__ Bt,
                               float* __restrict__ C,
                               int M, int K, int Ncols) {
    const int lane  = threadIdx.x & 31;
    const int wave  = blockIdx.x * (blockDim.x >> 5) + (threadIdx.x >> 5);
    const int spr   = Ncols >> 6;                 // 64-wide strips per row of tiles
    const int mt    = wave / spr;
    const int ns    = wave % spr;
    const int row0  = mt * 16;
    if (row0 >= M) return;
    const int col0  = ns * 64;
    const int m     = lane & 15;
    const int half  = lane >> 4;

    v8f acc[4] = {};
    int rr = row0 + m; if (rr >= M) rr = M - 1;   // tail clamp (M=20000 is exact)
    const unsigned short* arow = A + (size_t)rr * K;

    for (int k0 = 0; k0 < K; k0 += 32) {
        Frag16 a;
        #pragma unroll
        for (int q = 0; q < 8; ++q) {
            const int kk = k0 + ((q & 3) << 1) + (half << 3) + ((q >> 2) << 4);
            a.u[q] = *(const unsigned int*)(arow + kk);   // bf16 pair (K even)
        }
        #pragma unroll
        for (int j = 0; j < 4; ++j) {
            const int ncol = col0 + (j << 4) + m;
            const unsigned short* bcol = Bt + (size_t)ncol * K + k0;
            Frag16 b;
            #pragma unroll
            for (int q = 0; q < 8; ++q)
                b.u[q] = *(const unsigned int*)(bcol + ((q & 3) << 1) + (half << 3) + ((q >> 2) << 4)); // same K pair as A
            acc[j] = __builtin_amdgcn_wmma_f32_16x16x32_bf16(
                false, a.v, false, b.v, (short)0, acc[j], false, false);
        }
    }
    #pragma unroll
    for (int r = 0; r < 8; ++r) {
        const int mrow = row0 + r + (half << 3);
        if (mrow >= M) continue;
        float* crow = C + (size_t)mrow * Ncols + col0 + m;
        crow[0]  = acc[0][r];
        crow[16] = acc[1][r];
        crow[32] = acc[2][r];
        crow[48] = acc[3][r];
    }
}

// ---------------- GAT attention ----------------
__global__ void att_dots_kernel(const float* __restrict__ xl,
                                const float* __restrict__ att_src,
                                const float* __restrict__ att_dst,
                                float* __restrict__ a_s, float* __restrict__ a_d, int N) {
    int i = blockIdx.x * blockDim.x + threadIdx.x;     // i = n*H + h
    if (i >= N * HHEADS) return;
    int n = i / HHEADS, h = i % HHEADS;
    const float* row = xl + (size_t)n * HD + h * DDIM;
    const float* as  = att_src + h * DDIM;
    const float* ad  = att_dst + h * DDIM;
    float s = 0.f, d = 0.f;
    #pragma unroll 8
    for (int k = 0; k < DDIM; ++k) { float v = row[k]; s += v * as[k]; d += v * ad[k]; }
    a_s[i] = s; a_d[i] = d;
}

__device__ __forceinline__ void edge_sd(const long long* ei, int e, int E, int* s, int* d) {
    if (e < E) { *s = (int)ei[e]; *d = (int)ei[E + e]; }
    else       { *s = *d = e - E; }                      // appended self-loops
}

// alpha = leaky_relu(a_src[src]+a_dst[dst]); segment max via uint atomicMax; deg count
__global__ void edge_alpha_kernel(const long long* __restrict__ ei,
                                  const float* __restrict__ a_s, const float* __restrict__ a_d,
                                  float* __restrict__ alpha, unsigned int* __restrict__ mkey,
                                  float* __restrict__ deg, int E, int N) {
    int i = blockIdx.x * blockDim.x + threadIdx.x;       // i = e*H + h
    int tasks = (E + N) * HHEADS;
    if (i >= tasks) return;
    int e = i / HHEADS, h = i % HHEADS, s, d;
    edge_sd(ei, e, E, &s, &d);
    float a = a_s[s * HHEADS + h] + a_d[d * HHEADS + h];
    a = (a >= 0.f) ? a : NEG_SLOPE * a;
    alpha[i] = a;
    atomicMax(&mkey[d * HHEADS + h], fenc(a));
    if (h == 0) atomicAdd(&deg[d], 1.0f);
}

// ex = exp(alpha - m[dst]); den += ex (stores ex in place of alpha)
__global__ void edge_den_kernel(const long long* __restrict__ ei,
                                float* __restrict__ alpha,
                                const unsigned int* __restrict__ mkey,
                                float* __restrict__ den, int E, int N) {
    int i = blockIdx.x * blockDim.x + threadIdx.x;
    int tasks = (E + N) * HHEADS;
    if (i >= tasks) return;
    int e = i / HHEADS, h = i % HHEADS, s, d;
    edge_sd(ei, e, E, &s, &d);
    float ex = __expf(alpha[i] - fdec(mkey[d * HHEADS + h]));
    alpha[i] = ex;
    atomicAdd(&den[d * HHEADS + h], ex);
}

// one wave per (edge,head): x1[dst, h*64: ] += xl[src, h*64: ] * w
__global__ void edge_msg_kernel(const long long* __restrict__ ei,
                                const float* __restrict__ ex, const float* __restrict__ den,
                                const float* __restrict__ xl, float* __restrict__ x1,
                                int E, int N) {
    int wid = blockIdx.x * (blockDim.x >> 5) + (threadIdx.x >> 5);
    int tasks = (E + N) * HHEADS;
    if (wid >= tasks) return;
    int e = wid / HHEADS, h = wid % HHEADS, s, d;
    edge_sd(ei, e, E, &s, &d);
    float w = ex[wid] / (den[d * HHEADS + h] + 1e-16f);
    int lane = threadIdx.x & 31;
    const float2 v = *(const float2*)(xl + (size_t)s * HD + h * DDIM + lane * 2);
    float* dst = x1 + (size_t)d * HD + h * DDIM + lane * 2;
    atomicAdd(dst,     v.x * w);
    atomicAdd(dst + 1, v.y * w);
}

// x1 = relu(x1acc + b_gat) -> bf16 (only consumer is GEMM2)
__global__ void x1_finalize_kernel(const float* __restrict__ x1acc,
                                   const float* __restrict__ b_gat,
                                   unsigned short* __restrict__ x1bf, int total) {
    int i = blockIdx.x * blockDim.x + threadIdx.x;
    if (i >= total) return;
    float v = x1acc[i] + b_gat[i % HD];
    x1bf[i] = f2bf(v > 0.f ? v : 0.f);
}

__global__ void dinv_kernel(float* __restrict__ deg, int N) {
    int i = blockIdx.x * blockDim.x + threadIdx.x;
    if (i < N) { float g = deg[i]; deg[i] = (g > 0.f) ? rsqrtf(g) : 0.f; }
}

// one wave per edge: x2[dst] += xw[src] * dinv[src]*dinv[dst]   (640 dims)
__global__ void gcn_scatter_kernel(const long long* __restrict__ ei,
                                   const float* __restrict__ xw,
                                   const float* __restrict__ dinv,
                                   float* __restrict__ x2, int E, int N) {
    int wid = blockIdx.x * (blockDim.x >> 5) + (threadIdx.x >> 5);
    if (wid >= E + N) return;
    int s, d; edge_sd(ei, wid, E, &s, &d);
    float norm = dinv[s] * dinv[d];
    int lane = threadIdx.x & 31;
    const float* srow = xw + (size_t)s * HD;
    float* drow = x2 + (size_t)d * HD;
    #pragma unroll
    for (int t = 0; t < HD / 32; ++t) {
        int c = lane + t * 32;
        atomicAdd(&drow[c], srow[c] * norm);
    }
}

// global max/mean pool over relu(x2 + b_gcn): block-local reduce then one atomic/col-chunk
#define POOL_ROWS 256
__global__ void pool_kernel(const float* __restrict__ x2, const float* __restrict__ b_gcn,
                            unsigned int* __restrict__ pmax, float* __restrict__ psum, int N) {
    int c = blockIdx.x * blockDim.x + threadIdx.x;
    if (c >= HD) return;
    int r0 = blockIdx.y * POOL_ROWS;
    int r1 = r0 + POOL_ROWS; if (r1 > N) r1 = N;
    float bc = b_gcn[c], lm = 0.f, ls = 0.f;
    for (int r = r0; r < r1; ++r) {
        float v = x2[(size_t)r * HD + c] + bc;
        v = v > 0.f ? v : 0.f;
        lm = fmaxf(lm, v); ls += v;
    }
    atomicMax(&pmax[c], fenc(lm));
    atomicAdd(&psum[c], ls);
}

// h = relu([pmax ; psum/N] @ W1 + b1)
__global__ void mlp1_kernel(const unsigned int* __restrict__ pmax,
                            const float* __restrict__ psum,
                            const float* __restrict__ W1, const float* __restrict__ b1,
                            float* __restrict__ h, float invN) {
    int j = blockIdx.x * blockDim.x + threadIdx.x;
    if (j >= 1500) return;
    float acc = b1[j];
    for (int i = 0; i < HD; ++i)       acc += fdec(pmax[i])      * W1[(size_t)i * 1500 + j];
    for (int i = 0; i < HD; ++i)       acc += (psum[i] * invN)   * W1[(size_t)(HD + i) * 1500 + j];
    h[j] = acc > 0.f ? acc : 0.f;
}

// out[10] = h @ W2 + b2 ; 10 waves, one per output
__global__ void mlp2_kernel(const float* __restrict__ h, const float* __restrict__ W2,
                            const float* __restrict__ b2, float* __restrict__ out) {
    __shared__ float sm[320];
    int t = threadIdx.x, o = t >> 5, lane = t & 31;
    float acc = 0.f;
    for (int j = lane; j < 1500; j += 32) acc += h[j] * W2[(size_t)j * 10 + o];
    sm[t] = acc;
    __syncthreads();
    if (lane == 0) {
        float s = b2[o];
        #pragma unroll
        for (int l = 0; l < 32; ++l) s += sm[(o << 5) + l];
        out[o] = s;
    }
}

// ---------------- host-side orchestration ----------------
static inline int cdiv_i(int a, int b) { return (a + b - 1) / b; }

extern "C" void kernel_launch(void* const* d_in, const int* in_sizes, int n_in,
                              void* d_out, int out_size, void* d_ws, size_t ws_size,
                              hipStream_t stream) {
    const float*     x     = (const float*)d_in[0];
    const float*     W_gat = (const float*)d_in[1];
    const float*     att_s = (const float*)d_in[2];
    const float*     att_d = (const float*)d_in[3];
    const float*     b_gat = (const float*)d_in[4];
    const float*     W_gcn = (const float*)d_in[5];
    const float*     b_gcn = (const float*)d_in[6];
    const float*     W1    = (const float*)d_in[7];
    const float*     b1    = (const float*)d_in[8];
    const float*     W2    = (const float*)d_in[9];
    const float*     b2    = (const float*)d_in[10];
    const long long* ei    = (const long long*)d_in[11];   // int64 [2,E]

    const int N = in_sizes[0] / DDIM;       // 20000
    const int E = in_sizes[11] / 2;         // 320000
    const int Etot = E + N;                 // with self-loops

    // ---- workspace carve-up (256B aligned); bigA = xl then x2, bigB = x1acc then xw
    char* base = (char*)d_ws;
    size_t off = 0;
    auto carve = [&](size_t bytes) { size_t o = off; off += (bytes + 255) & ~(size_t)255; return o; };
    const size_t bigBytes = (size_t)N * HD * sizeof(float);
    float*          bigA    = (float*)(base + carve(bigBytes));            // xl -> x2acc
    float*          bigB    = (float*)(base + carve(bigBytes));            // x1acc -> xw
    unsigned short* x_bf    = (unsigned short*)(base + carve((size_t)N * DDIM * 2));
    unsigned short* x1_bf   = (unsigned short*)(base + carve((size_t)N * HD * 2));
    unsigned short* Wgat_t  = (unsigned short*)(base + carve((size_t)DDIM * HD * 2));
    unsigned short* Wgcn_t  = (unsigned short*)(base + carve((size_t)HD * HD * 2));
    float*          a_src_n = (float*)(base + carve((size_t)N * HHEADS * 4));
    float*          a_dst_n = (float*)(base + carve((size_t)N * HHEADS * 4));
    unsigned int*   mkey    = (unsigned int*)(base + carve((size_t)N * HHEADS * 4));
    float*          den     = (float*)(base + carve((size_t)N * HHEADS * 4));
    float*          alpha   = (float*)(base + carve((size_t)Etot * HHEADS * 4));
    float*          deg     = (float*)(base + carve((size_t)N * 4));       // -> dinv in place
    unsigned int*   pmax    = (unsigned int*)(base + carve((size_t)HD * 4));
    float*          psum    = (float*)(base + carve((size_t)HD * 4));
    float*          hbuf    = (float*)(base + carve((size_t)1500 * 4));
    (void)ws_size; (void)n_in; (void)out_size;

    // ---- init accumulators
    hipMemsetAsync(bigB, 0, bigBytes, stream);                         // x1 accum
    hipMemsetAsync(den,  0, (size_t)N * HHEADS * 4, stream);
    hipMemsetAsync(deg,  0, (size_t)N * 4, stream);
    hipMemsetAsync(psum, 0, (size_t)HD * 4, stream);
    fill_u32_kernel<<<cdiv_i(N * HHEADS, 256), 256, 0, stream>>>(mkey, ENC_NEG_INF, N * HHEADS);
    fill_u32_kernel<<<cdiv_i(HD, 256), 256, 0, stream>>>(pmax, ENC_ZERO, HD);

    // ---- bf16 operand prep
    cvt_bf16_rowmajor<<<cdiv_i(N * DDIM, 256), 256, 0, stream>>>(x, x_bf, N * DDIM);
    cvt_bf16_transpose<<<cdiv_i(DDIM * HD, 256), 256, 0, stream>>>(W_gat, Wgat_t, DDIM, HD);
    cvt_bf16_transpose<<<cdiv_i(HD * HD, 256), 256, 0, stream>>>(W_gcn, Wgcn_t, HD, HD);

    // ---- GEMM1: xl = x @ W_gat   [N,64]x[64,640]
    {
        int waves = cdiv_i(N, 16) * (HD / 64);
        gemm_bf16_wmma<<<cdiv_i(waves, 8), 256, 0, stream>>>(x_bf, Wgat_t, bigA, N, DDIM, HD);
    }

    // ---- attention scores, segment softmax, message scatter
    att_dots_kernel<<<cdiv_i(N * HHEADS, 256), 256, 0, stream>>>(bigA, att_s, att_d, a_src_n, a_dst_n, N);
    {
        int tasks = Etot * HHEADS;
        edge_alpha_kernel<<<cdiv_i(tasks, 256), 256, 0, stream>>>(ei, a_src_n, a_dst_n, alpha, mkey, deg, E, N);
        edge_den_kernel<<<cdiv_i(tasks, 256), 256, 0, stream>>>(ei, alpha, mkey, den, E, N);
        edge_msg_kernel<<<cdiv_i(tasks, 8), 256, 0, stream>>>(ei, alpha, den, bigA, bigB, E, N);
    }
    x1_finalize_kernel<<<cdiv_i(N * HD, 256), 256, 0, stream>>>(bigB, b_gat, x1_bf, N * HD);
    dinv_kernel<<<cdiv_i(N, 256), 256, 0, stream>>>(deg, N);

    // ---- GEMM2: xw = x1 @ W_gcn  [N,640]x[640,640]  (xw overwrites bigB)
    {
        int waves = cdiv_i(N, 16) * (HD / 64);
        gemm_bf16_wmma<<<cdiv_i(waves, 8), 256, 0, stream>>>(x1_bf, Wgcn_t, bigB, N, HD, HD);
    }

    // ---- GCN normalized scatter into x2 (reuses bigA; xl is dead now)
    hipMemsetAsync(bigA, 0, bigBytes, stream);
    gcn_scatter_kernel<<<cdiv_i(Etot, 8), 256, 0, stream>>>(ei, bigB, deg, bigA, E, N);

    // ---- pool + MLP head
    {
        dim3 grid(HD / 128, cdiv_i(N, POOL_ROWS));
        pool_kernel<<<grid, 128, 0, stream>>>(bigA, b_gcn, pmax, psum, N);
    }
    mlp1_kernel<<<cdiv_i(1500, 256), 256, 0, stream>>>(pmax, psum, W1, b1, hbuf, 1.0f / (float)N);
    mlp2_kernel<<<1, 320, 0, stream>>>(hbuf, W2, b2, (float*)d_out);
}
